// GraphGNN_618475290909
// MI455X (gfx1250) — matrix operand order repaired and checked by reading, not compiled, not run on hardware
//
#include <hip/hip_runtime.h>

// ---------------------------------------------------------------------------
// GNN edge-MLP + scatter + node-MLP for MI455X (gfx1250, wave32, WMMA).
// x (25.6MB) and the (B,N,30) accumulator (24MB) both fit in the 192MB L2,
// so the edge phase is L2-resident; the MLPs run at WMMA f16 rate and the
// scatter uses native global_atomic_add_f32. B operands are pre-swizzled
// into fragment order (ds_load_b128), conversions use v_cvt_pk_rtz_f16_f32,
// sigmoid uses the hardware v_exp_f32 + v_rcp_f32 transcendentals.
// ---------------------------------------------------------------------------

typedef _Float16 half_t;
typedef __attribute__((ext_vector_type(4)))  _Float16 v4h;
typedef __attribute__((ext_vector_type(8)))  _Float16 v8h;
typedef __attribute__((ext_vector_type(16))) _Float16 v16h;
typedef __attribute__((ext_vector_type(8)))  float    v8f;
typedef __attribute__((ext_vector_type(2)))  __fp16   f16x2;
typedef __attribute__((ext_vector_type(4)))  __fp16   f16x4;

#define WMMA_F32_F16(A, B, C) \
  __builtin_amdgcn_wmma_f32_16x16x32_f16(false, (A), false, (B), (short)0, (C), false, false)

// sigmoid via v_exp_f32 + v_rcp_f32 (4 VALU ops, no IEEE-div expansion)
__device__ __forceinline__ float sigmoidf_(float v) {
  return __builtin_amdgcn_rcpf(1.0f + __expf(-v));
}

// A-matrix 16x32 f16 fragment: lane L holds row r=L&15;
// lanes 0-15: K {0..15}, lanes 16-31: K {16..31}.
// Both 8-half chunks are 16B-contiguous -> two ds_load_b128.
__device__ __forceinline__ v16h load_a_16x32(const half_t* m, int lane) {
  const int r = lane & 15, hi = lane >> 4;
  v8h lo = *(const v8h*)(m + r * 32 + 16 * hi);
  v8h up = *(const v8h*)(m + r * 32 + 16 * hi + 8);
  return __builtin_shufflevector(lo, up, 0, 1, 2, 3, 4, 5, 6, 7,
                                 8, 9, 10, 11, 12, 13, 14, 15);
}

// B fragments are pre-swizzled in LDS: fB[(tile*32 + lane)*16 + j] where
// j=0..15 are the lane's 16 K-values (col = 16*tile + (lane&15),
// K = 16*(lane>>4)+j). 32B contiguous per lane -> two ds_load_b128.
__device__ __forceinline__ v16h load_b_frag(const half_t* fB, int tile, int lane) {
  const half_t* p = fB + (tile * 32 + lane) * 16;
  v8h lo = *(const v8h*)p;
  v8h up = *(const v8h*)(p + 8);
  return __builtin_shufflevector(lo, up, 0, 1, 2, 3, 4, 5, 6, 7,
                                 8, 9, 10, 11, 12, 13, 14, 15);
}

// pack 4 f32 -> 4 f16 with v_cvt_pk_rtz_f16_f32
__device__ __forceinline__ v4h pack4(float a, float b, float c, float d) {
  f16x2 p0 = __builtin_amdgcn_cvt_pkrtz(a, b);
  f16x2 p1 = __builtin_amdgcn_cvt_pkrtz(c, d);
  f16x4 p = __builtin_shufflevector(p0, p1, 0, 1, 2, 3);
  return __builtin_bit_cast(v4h, p);
}

// ---------------------------------------------------------------------------
__global__ __launch_bounds__(256) void gnn_zero_kernel(float* p, int n) {
  int i = blockIdx.x * 256 + threadIdx.x;
  if (i < n) p[i] = 0.0f;
}

// sum & sum-of-squares of edge_attr -> ws[0], ws[1]
__global__ __launch_bounds__(256) void gnn_stats_kernel(const float* __restrict__ ea,
                                                        int E, float* __restrict__ ws) {
  __shared__ float ssum[256], ssq[256];
  const int tid = threadIdx.x;
  float s = 0.0f, q = 0.0f;
  for (int i = blockIdx.x * 256 + tid; i < E; i += gridDim.x * 256) {
    float v = ea[i];
    s += v;
    q += v * v;
  }
  ssum[tid] = s; ssq[tid] = q;
  __syncthreads();
  for (int off = 128; off > 0; off >>= 1) {
    if (tid < off) { ssum[tid] += ssum[tid + off]; ssq[tid] += ssq[tid + off]; }
    __syncthreads();
  }
  if (tid == 0) {
    unsafeAtomicAdd(&ws[0], ssum[0]);
    unsafeAtomicAdd(&ws[1], ssq[0]);
  }
}

// ---------------------------------------------------------------------------
// Edge kernel: block = 128 threads = 4 waves; wave handles 16 edges of one
// batch. blockIdx.x tiles 64 edges, blockIdx.y = batch.
__global__ __launch_bounds__(128) void gnn_edge_kernel(
    const float* __restrict__ x, const int* __restrict__ ei,
    const float* __restrict__ ea, const float* __restrict__ W1,
    const float* __restrict__ b1, const float* __restrict__ W2,
    const float* __restrict__ b2, float* __restrict__ acc,
    const float* __restrict__ stats, int N, int E)
{
  // weights, pre-swizzled into B-fragment order: [2 tiles][32 lanes][16]
  __shared__ half_t w1sF[2048];
  __shared__ half_t w1tF[2048];
  __shared__ half_t w2pF[2048];
  __shared__ float  w1e_s[32], b1_s[32], b2_s[32];
  // per-wave tiles
  __shared__ half_t Asrc[4][16 * 32];
  __shared__ half_t Atgt[4][16 * 32];
  __shared__ half_t H1[4][16 * 32];
  __shared__ __attribute__((aligned(16))) float ea_s[4][16];
  __shared__ __attribute__((aligned(16))) int   src_s[4][16];
  __shared__ __attribute__((aligned(16))) int   tgt_s[4][16];

  const int tid  = threadIdx.x;
  const int lane = tid & 31;
  const int w    = tid >> 5;
  const int b    = blockIdx.y;

  // stage weights directly in fragment order (f32 -> f16)
  for (int i = tid; i < 2048; i += 128) {
    const int tl = i >> 10, rest = i & 1023;
    const int L = rest >> 4, j = rest & 15;
    const int k = 16 * (L >> 4) + j;        // K index
    const int n = 16 * tl + (L & 15);       // N index (output column)
    w1sF[i] = (half_t)W1[k * 32 + n];
    w1tF[i] = (half_t)W1[(32 + k) * 32 + n];
    w2pF[i] = (half_t)((n < 30) ? W2[k * 30 + n] : 0.0f);
  }
  if (tid < 32) {
    w1e_s[tid] = W1[64 * 32 + tid];          // ea rank-1 row of W1
    b1_s[tid]  = b1[tid];
    b2_s[tid]  = (tid < 30) ? b2[tid] : 0.0f;
  }

  // edge_attr normalization constants (ddof=1)
  const float sum = stats[0], sq = stats[1];
  const float mean = sum / (float)E;
  const float inv_std = rsqrtf((sq - sum * mean) / (float)(E - 1));

  const int eBase = blockIdx.x * 64 + w * 16;
  __builtin_prefetch(ei + eBase, 0, 0);       // global_prefetch_b8

  if (lane < 16) {
    int e = eBase + lane;
    int s = 0, t = 0; float v = 0.0f;
    if (e < E) { s = ei[e]; t = ei[E + e]; v = (ea[e] - mean) * inv_std; }
    src_s[w][lane] = s; tgt_s[w][lane] = t; ea_s[w][lane] = v;
  }
  __syncthreads();

  // gather: 16 rows x (32 src + 32 tgt) floats; one b128 global load +
  // one b64 LDS store per side per lane-iteration
#pragma unroll
  for (int it = 0; it < 4; ++it) {
    int idx = it * 32 + lane;          // 0..127
    int r = idx >> 3, q = idx & 7;     // row, 16B quad
    const float4 vs = ((const float4*)(x + ((size_t)b * N + src_s[w][r]) * 32))[q];
    const float4 vt = ((const float4*)(x + ((size_t)b * N + tgt_s[w][r]) * 32))[q];
    *(v4h*)&Asrc[w][r * 32 + q * 4] = pack4(vs.x, vs.y, vs.z, vs.w);
    *(v4h*)&Atgt[w][r * 32 + q * 4] = pack4(vt.x, vt.y, vt.z, vt.w);
  }
  __syncthreads();

  const int hi = lane >> 4;
  const v16h a_s = load_a_16x32(Asrc[w], lane);
  const v16h a_t = load_a_16x32(Atgt[w], lane);

  // hoist per-row epilogue operands into registers (b128 LDS loads)
  float ear[8];
  {
    const float4 e0 = *(const float4*)&ea_s[w][8 * hi];
    const float4 e1 = *(const float4*)&ea_s[w][8 * hi + 4];
    ear[0] = e0.x; ear[1] = e0.y; ear[2] = e0.z; ear[3] = e0.w;
    ear[4] = e1.x; ear[5] = e1.y; ear[6] = e1.z; ear[7] = e1.w;
  }
  int tg[8], sc[8];
  {
    const int4 t0 = *(const int4*)&tgt_s[w][8 * hi];
    const int4 t1 = *(const int4*)&tgt_s[w][8 * hi + 4];
    tg[0] = t0.x; tg[1] = t0.y; tg[2] = t0.z; tg[3] = t0.w;
    tg[4] = t1.x; tg[5] = t1.y; tg[6] = t1.z; tg[7] = t1.w;
    const int4 s0 = *(const int4*)&src_s[w][8 * hi];
    const int4 s1 = *(const int4*)&src_s[w][8 * hi + 4];
    sc[0] = s0.x; sc[1] = s0.y; sc[2] = s0.z; sc[3] = s0.w;
    sc[4] = s1.x; sc[5] = s1.y; sc[6] = s1.z; sc[7] = s1.w;
  }

  // layer 1: h1 = sigmoid(xs@W1s + xt@W1t + ea*w1e + b1)   (16x32)
#pragma unroll
  for (int tl = 0; tl < 2; ++tl) {
    v16h bs = load_b_frag(w1sF, tl, lane);
    v16h bt = load_b_frag(w1tF, tl, lane);
    v8f c = {};
    c = WMMA_F32_F16(a_s, bs, c);
    c = WMMA_F32_F16(a_t, bt, c);
    const int col = 16 * tl + (lane & 15);
    const float we = w1e_s[col], bb = b1_s[col];
#pragma unroll
    for (int i = 0; i < 8; ++i) {
      float v = c[i] + ear[i] * we + bb;
      H1[w][(i + 8 * hi) * 32 + col] = (half_t)sigmoidf_(v);
    }
  }
  __syncthreads();

  // layer 2: h2 = sigmoid(h1@W2 + b2)  (16x30), then scatter +tgt / -src
  const v16h a_h = load_a_16x32(H1[w], lane);
  const size_t rowOff = (size_t)b * N;
#pragma unroll
  for (int tl = 0; tl < 2; ++tl) {
    v16h bv = load_b_frag(w2pF, tl, lane);
    v8f c = {};
    c = WMMA_F32_F16(a_h, bv, c);
    const int col = 16 * tl + (lane & 15);
    if (col < 30) {
      const float bb = b2_s[col];
#pragma unroll
      for (int i = 0; i < 8; ++i) {
        if (eBase + i + 8 * hi < E) {
          float h2 = sigmoidf_(c[i] + bb);
          unsafeAtomicAdd(acc + (rowOff + tg[i]) * 30 + col,  h2);
          unsafeAtomicAdd(acc + (rowOff + sc[i]) * 30 + col, -h2);
        }
      }
    }
  }
}

// ---------------------------------------------------------------------------
// Node kernel: out = sigmoid(acc @ W3 + b3), rows = B*N, K padded 30->32.
__global__ __launch_bounds__(128) void gnn_node_kernel(
    const float* __restrict__ acc, const float* __restrict__ W3,
    const float* __restrict__ b3, float* __restrict__ out, int rows)
{
  __shared__ half_t w3pF[2048];         // W3 in fragment order (K>=30 -> 0)
  __shared__ float  b3_s[32];
  __shared__ half_t Arow[4][16 * 32];

  const int tid = threadIdx.x, lane = tid & 31, w = tid >> 5;
  for (int i = tid; i < 2048; i += 128) {
    const int tl = i >> 10, rest = i & 1023;
    const int L = rest >> 4, j = rest & 15;
    const int k = 16 * (L >> 4) + j;
    const int n = 16 * tl + (L & 15);
    w3pF[i] = (half_t)((k < 30) ? W3[k * 32 + n] : 0.0f);
  }
  if (tid < 32) b3_s[tid] = b3[tid];

  const int rowBase = blockIdx.x * 64 + w * 16;
#pragma unroll
  for (int it = 0; it < 16; ++it) {
    int idx = it * 32 + lane;          // 0..511
    int r = idx >> 5, k = idx & 31;
    float v = 0.0f;
    if (k < 30 && rowBase + r < rows) v = acc[(size_t)(rowBase + r) * 30 + k];
    Arow[w][r * 32 + k] = (half_t)v;
  }
  __syncthreads();

  const v16h a  = load_a_16x32(Arow[w], lane);
  const int  hi = lane >> 4;
#pragma unroll
  for (int tl = 0; tl < 2; ++tl) {
    v16h bv = load_b_frag(w3pF, tl, lane);
    v8f c = {};
    c = WMMA_F32_F16(a, bv, c);
    const int col = 16 * tl + (lane & 15);
    const float bb = b3_s[col];
#pragma unroll
    for (int i = 0; i < 8; ++i) {
      int rr = rowBase + i + 8 * hi;
      if (rr < rows) out[(size_t)rr * 32 + col] = sigmoidf_(c[i] + bb);
    }
  }
}

// ---------------------------------------------------------------------------
extern "C" void kernel_launch(void* const* d_in, const int* in_sizes, int n_in,
                              void* d_out, int out_size, void* d_ws, size_t ws_size,
                              hipStream_t stream) {
  const float* x   = (const float*)d_in[0];
  const int*   ei  = (const int*)d_in[1];     // (2,E) flat: [src | tgt]
  const float* ea  = (const float*)d_in[2];
  const float* W1  = (const float*)d_in[3];
  const float* b1  = (const float*)d_in[4];
  const float* W2  = (const float*)d_in[5];
  const float* b2  = (const float*)d_in[6];
  const float* W3  = (const float*)d_in[7];
  const float* b3  = (const float*)d_in[8];
  float*       out = (float*)d_out;
  float*       wsf = (float*)d_ws;

  const int B = 4;
  const int E = in_sizes[2];                  // 800000
  const int N = in_sizes[0] / (B * 32);       // 50000
  const int rows = B * N;                     // 200000

  float* stats = wsf;                         // [0]=sum, [1]=sumsq
  float* acc   = wsf + 64;                    // (B*N, 30) accumulator, 24MB

  // 1) zero stats + accumulator (ws is poisoned between runs)
  const int nz = 64 + rows * 30;
  gnn_zero_kernel<<<(nz + 255) / 256, 256, 0, stream>>>(wsf, nz);

  // 2) edge_attr mean/std reduction
  gnn_stats_kernel<<<256, 256, 0, stream>>>(ea, E, stats);

  // 3) edge MLP + scatter (grid: 64 edges per block x B batches)
  dim3 egrid((E + 63) / 64, B);
  gnn_edge_kernel<<<egrid, 128, 0, stream>>>(x, ei, ea, W1, b1, W2, b2,
                                             acc, stats, N, E);

  // 4) node MLP
  gnn_node_kernel<<<(rows + 63) / 64, 128, 0, stream>>>(acc, W3, b3, out, rows);
}
